// ContrastCELoss_34359739047
// MI455X (gfx1250) — compile-verified, matrix operand order checked by reading, not run on hardware
//
#include <hip/hip_runtime.h>
#include <hip/hip_bf16.h>
#include <stdint.h>

#define MAX_SAMPLES 1024
#define MAX_VIEWS_C 100
#define TEMP        0.07f
#define BASE_TEMP   0.07f
#define LOSS_WEIGHT 0.1f
#define MMAX        1024   // padded GEMM size (>= T*n_view <= 1024)
#define KDIM        128    // feature dim D
#define LDL         1024   // logits leading dimension

typedef __attribute__((ext_vector_type(16))) __bf16 v16bf;
typedef __attribute__((ext_vector_type(8)))  __bf16 v8bf;
typedef __attribute__((ext_vector_type(8)))  float  v8f;

// ---------------------------------------------------------------- k0: zero ints
__global__ void k0_zero_i32(int* p, int n) {
    int i = blockIdx.x * 256 + threadIdx.x;
    if (i < n) p[i] = 0;
}

// ------------------------------------------- k1: seg partials + argmax + counts
__global__ void k1_seg_pred(const float* __restrict__ outputs,
                            const long long* __restrict__ labels,
                            const float* __restrict__ loss_w,
                            int C, int HW, int BN,
                            int* __restrict__ pred_buf,
                            int* __restrict__ cnt_tot,
                            int* __restrict__ cnt_hard,
                            float* __restrict__ partial) {
    __shared__ float s0[256], s1[256];
    int g = blockIdx.x * 256 + threadIdx.x;
    float wlp = 0.0f, wsum = 0.0f;
    if (g < BN) {
        int b   = g / HW;
        int pix = g - b * HW;
        int lab = (int)labels[g];
        const float* op = outputs + (size_t)b * C * HW + pix;
        float best = op[0]; int bi = 0;
        for (int c = 1; c < C; ++c) {
            float v = op[(size_t)c * HW];
            if (v > best) { best = v; bi = c; }   // first-max, matches jnp.argmax
        }
        pred_buf[g] = bi;
        float w  = loss_w[lab];
        float lp = op[(size_t)lab * HW];
        wlp = w * lp; wsum = w;
        atomicAdd(&cnt_tot[b * C + lab], 1);
        if (bi != lab) atomicAdd(&cnt_hard[b * C + lab], 1);
    }
    s0[threadIdx.x] = wlp; s1[threadIdx.x] = wsum;
    __syncthreads();
    for (int s = 128; s > 0; s >>= 1) {
        if (threadIdx.x < s) {
            s0[threadIdx.x] += s0[threadIdx.x + s];
            s1[threadIdx.x] += s1[threadIdx.x + s];
        }
        __syncthreads();
    }
    if (threadIdx.x == 0) {
        partial[2 * blockIdx.x]     = s0[0];
        partial[2 * blockIdx.x + 1] = s1[0];
    }
}

// -------------------------- k2: deterministic seg reduce + sampling plan (slots)
__global__ void k2_plan(const int* __restrict__ cnt_tot,
                        const int* __restrict__ cnt_hard,
                        int C, int B, int nb1,
                        const float* __restrict__ partial,
                        int* __restrict__ meta,      // [0]=T [1]=n_view [2]=M
                        int* __restrict__ slot_b, int* __restrict__ slot_c,
                        int* __restrict__ slot_kh, int* __restrict__ slot_ke,
                        float* __restrict__ segvals) {
    __shared__ float s0[256], s1[256];
    float a0 = 0.0f, a1 = 0.0f;
    for (int i = threadIdx.x; i < nb1; i += 256) {
        a0 += partial[2 * i];
        a1 += partial[2 * i + 1];
    }
    s0[threadIdx.x] = a0; s1[threadIdx.x] = a1;
    __syncthreads();
    for (int s = 128; s > 0; s >>= 1) {
        if (threadIdx.x < s) {
            s0[threadIdx.x] += s0[threadIdx.x + s];
            s1[threadIdx.x] += s1[threadIdx.x + s];
        }
        __syncthreads();
    }
    if (threadIdx.x == 0) {
        segvals[0] = -(s0[0] / s1[0]);            // seg_loss
        int total = 0;
        for (int b = 0; b < B; ++b)
            for (int c = 1; c < C; ++c)           // c==0 is IGNORE
                if (cnt_tot[b * C + c] > MAX_VIEWS_C) total++;
        int nv = (total > 0) ? (MAX_SAMPLES / total) : 1;
        if (nv > MAX_VIEWS_C) nv = MAX_VIEWS_C;
        int ptr = 0;
        for (int b = 0; b < B; ++b) {
            for (int c = 1; c < C; ++c) {
                int ct = cnt_tot[b * C + c];
                if (ct > MAX_VIEWS_C) {
                    int nh = cnt_hard[b * C + c];
                    int ne = ct - nh;
                    int kh, ke;
                    if (2 * nh >= nv && 2 * ne >= nv) { kh = nv / 2;  ke = nv - kh; }
                    else if (2 * nh >= nv)            { ke = ne;      kh = nv - ke; }
                    else                              { kh = nh;      ke = nv - kh; }
                    slot_b[ptr] = b; slot_c[ptr] = c;
                    slot_kh[ptr] = kh; slot_ke[ptr] = ke;
                    ptr++;
                }
            }
        }
        meta[0] = total;
        meta[1] = nv;
        meta[2] = total * nv;
    }
}

// ------------- k3: deterministic hard/easy pixel selection, one wave32 per slot
__global__ void k3_select(const long long* __restrict__ labels,
                          const int* __restrict__ pred_buf,
                          int HW,
                          const int* __restrict__ meta,
                          const int* __restrict__ slot_b, const int* __restrict__ slot_c,
                          const int* __restrict__ slot_kh, const int* __restrict__ slot_ke,
                          int* __restrict__ sample_pix) {
    int t = blockIdx.x;
    int T = meta[0];
    if (t >= T) return;
    int b = slot_b[t], c = slot_c[t], kh = slot_kh[t], ke = slot_ke[t];
    const long long* lb = labels   + (size_t)b * HW;
    const int*       pr = pred_buf + (size_t)b * HW;
    int lane = threadIdx.x;
    unsigned lanebits = (1u << lane) - 1u;
    int* outp = sample_pix + t * MAX_VIEWS_C;
    int th = 0, te = 0;
    for (int base = 0; base < HW; base += 32) {
        if (th >= kh && te >= ke) break;
        int pix = base + lane;                    // HW is a multiple of 32
        bool match = ((int)lb[pix] == c);
        bool hard  = match && (pr[pix] != c);
        bool easy  = match && (pr[pix] == c);
        unsigned mh = __builtin_amdgcn_ballot_w32(hard);
        unsigned me = __builtin_amdgcn_ballot_w32(easy);
        int rh = th + __popc(mh & lanebits);
        int re = te + __popc(me & lanebits);
        if (hard && rh < kh) outp[rh]      = pix;
        if (easy && re < ke) outp[kh + re] = pix;
        th += __popc(mh);
        te += __popc(me);
    }
}

// ------------------------- k4: gather sampled features -> bf16 [MMAX x KDIM]
// feat row m = v*T + t  (reference: transpose(X_, (1,0,2)).reshape(T*nv, d))
__global__ void k4_gather(const float* __restrict__ feats,
                          int Dch, int HW,
                          const int* __restrict__ meta,
                          const int* __restrict__ slot_b,
                          const int* __restrict__ sample_pix,
                          __bf16* __restrict__ Xbf) {
    int m = blockIdx.x;           // 0..MMAX-1
    int d = threadIdx.x;          // 0..KDIM-1
    int T = meta[0], M = meta[2];
    __bf16 out = (__bf16)0.0f;    // zero-pad rows >= M and channels >= Dch
    if (m < M && d < Dch) {
        int t = m % T, v = m / T;
        int b = slot_b[t];
        int pix = sample_pix[t * MAX_VIEWS_C + v];
        out = (__bf16)feats[((size_t)b * Dch + d) * HW + pix];
    }
    Xbf[(size_t)m * KDIM + d] = out;
}

// --------------------------- k5: logits = (X * X^T) / TEMP via bf16 WMMA
// One wave per 16x16 output tile; K=128 in four 16x16x32 WMMA steps.
// 16-bit A layout (ISA 7.12.2): lane holds row (lane&15); VGPR v holds K pairs
// k = 16*(e>>3) + 8*(lane>>4) + (e&7)  -> two contiguous 16B loads per K-step.
// For A*A^T the B fragment (32x16, column n = lane&15) loads identically from
// the j-block rows of X.
__global__ void __launch_bounds__(32)
k5_gemm(const __bf16* __restrict__ X, float* __restrict__ logits) {
    int lane = threadIdx.x;
    int r    = lane & 15;
    int half = lane >> 4;
    int ib = blockIdx.x * 16;
    int jb = blockIdx.y * 16;
    const __bf16* arow = X + (size_t)(ib + r) * KDIM + 8 * half;
    const __bf16* brow = X + (size_t)(jb + r) * KDIM + 8 * half;
    v8f acc = {};
#pragma unroll
    for (int kk = 0; kk < KDIM; kk += 32) {
        v8bf alo = *(const v8bf*)(arow + kk);
        v8bf ahi = *(const v8bf*)(arow + kk + 16);
        v8bf blo = *(const v8bf*)(brow + kk);
        v8bf bhi = *(const v8bf*)(brow + kk + 16);
        v16bf a, b;
#pragma unroll
        for (int i = 0; i < 8; ++i) {
            a[i] = alo[i]; a[i + 8] = ahi[i];
            b[i] = blo[i]; b[i + 8] = bhi[i];
        }
        acc = __builtin_amdgcn_wmma_f32_16x16x32_bf16(
            /*neg_a=*/false, a, /*neg_b=*/false, b,
            /*c_mod=*/(short)0, acc, /*reuse_a=*/false, /*reuse_b=*/false);
    }
    const float inv_temp = 1.0f / TEMP;
#pragma unroll
    for (int vr = 0; vr < 8; ++vr) {
        int row = ib + vr + 8 * half;             // C/D layout: M = vr + 8*half
        logits[(size_t)row * LDL + (jb + r)] = acc[vr] * inv_temp;
    }
}

// --------------- k6: per-row softmax-style passes (max / neg-sum / masked pos)
__global__ void k6_row(const float* __restrict__ logits,
                       const int* __restrict__ meta,
                       const int* __restrict__ slot_c,
                       float* __restrict__ rowvals) {
    int row = blockIdx.x;
    int T = meta[0], M = meta[2];
    if (row >= M) return;
    __shared__ float sred[256];
    __shared__ int   scls[64];
    int tid = threadIdx.x;
    if (tid < T && tid < 64) scls[tid] = slot_c[tid];
    __syncthreads();
    const float* L = logits + (size_t)row * LDL;
    int cr = scls[row % T];
    // pass 1: row max
    float mx = -3.4e38f;
    for (int j = tid; j < M; j += 256) mx = fmaxf(mx, L[j]);
    sred[tid] = mx; __syncthreads();
    for (int s = 128; s > 0; s >>= 1) {
        if (tid < s) sred[tid] = fmaxf(sred[tid], sred[tid + s]);
        __syncthreads();
    }
    float rowmax = sred[0]; __syncthreads();
    // pass 2: neg_logits = sum over different-class cols of exp(logit - max)
    float neg = 0.0f;
    for (int j = tid; j < M; j += 256)
        if (scls[j % T] != cr) neg += expf(L[j] - rowmax);
    sred[tid] = neg; __syncthreads();
    for (int s = 128; s > 0; s >>= 1) {
        if (tid < s) sred[tid] += sred[tid + s];
        __syncthreads();
    }
    float negsum = sred[0]; __syncthreads();
    // pass 3: sum of mask * log_prob and mask count (same class, no diagonal)
    float pos = 0.0f, cnt = 0.0f;
    for (int j = tid; j < M; j += 256) {
        if (j != row && scls[j % T] == cr) {
            float l = L[j] - rowmax;
            pos += l - logf(expf(l) + negsum);
            cnt += 1.0f;
        }
    }
    sred[tid] = pos; __syncthreads();
    for (int s = 128; s > 0; s >>= 1) {
        if (tid < s) sred[tid] += sred[tid + s];
        __syncthreads();
    }
    float possum = sred[0]; __syncthreads();
    sred[tid] = cnt; __syncthreads();
    for (int s = 128; s > 0; s >>= 1) {
        if (tid < s) sred[tid] += sred[tid + s];
        __syncthreads();
    }
    if (tid == 0) rowvals[row] = possum / sred[0];
}

// -------------------------------- k7: final fixed-order reduce and combination
__global__ void k7_final(const float* __restrict__ rowvals,
                         const int* __restrict__ meta,
                         const float* __restrict__ segvals,
                         float* __restrict__ out) {
    __shared__ float sred[256];
    int M = meta[2];
    float a = 0.0f;
    for (int j = threadIdx.x; j < M; j += 256) a += rowvals[j];
    sred[threadIdx.x] = a; __syncthreads();
    for (int s = 128; s > 0; s >>= 1) {
        if (threadIdx.x < s) sred[threadIdx.x] += sred[threadIdx.x + s];
        __syncthreads();
    }
    if (threadIdx.x == 0) {
        float contrast = -(TEMP / BASE_TEMP) * (sred[0] / (float)M);
        out[0] = segvals[0] + LOSS_WEIGHT * contrast;
    }
}

// ------------------------------------------------------------------ host entry
extern "C" void kernel_launch(void* const* d_in, const int* in_sizes, int n_in,
                              void* d_out, int out_size, void* d_ws, size_t ws_size,
                              hipStream_t stream) {
    const float*     feats   = (const float*)d_in[0];     // [B, D, H, W] f32
    const float*     outputs = (const float*)d_in[1];     // [B, C, H, W] f32
    const long long* labels  = (const long long*)d_in[2]; // [B, H, W] int64
    const float*     loss_w  = (const float*)d_in[3];     // [C] f32

    const int BN  = in_sizes[2];        // B*H*W
    const int C   = in_sizes[3];        // 20
    const int B   = 2;
    const int HW  = BN / B;
    const int Dch = in_sizes[0] / BN;   // 128
    const int TMAX = B * C;
    const int nb1  = (BN + 255) / 256;

    // ---- workspace layout
    uintptr_t base = (uintptr_t)d_ws;
    uintptr_t off  = 0;
    auto alignup = [](uintptr_t x, uintptr_t a) { return (x + a - 1) & ~(a - 1); };

    int* meta       = (int*)(base + off); off += 16 * sizeof(int);
    int* cnt_tot    = (int*)(base + off); off += (size_t)TMAX * sizeof(int);
    int* cnt_hard   = (int*)(base + off); off += (size_t)TMAX * sizeof(int);
    int* slot_b     = (int*)(base + off); off += (size_t)TMAX * sizeof(int);
    int* slot_c     = (int*)(base + off); off += (size_t)TMAX * sizeof(int);
    int* slot_kh    = (int*)(base + off); off += (size_t)TMAX * sizeof(int);
    int* slot_ke    = (int*)(base + off); off += (size_t)TMAX * sizeof(int);
    int* sample_pix = (int*)(base + off); off += (size_t)TMAX * MAX_VIEWS_C * sizeof(int);
    int* pred_buf   = (int*)(base + off); off += (size_t)BN * sizeof(int);
    float* partial  = (float*)(base + off); off += (size_t)2 * nb1 * sizeof(float);
    float* rowvals  = (float*)(base + off); off += (size_t)MMAX * sizeof(float);
    float* segvals  = (float*)(base + off); off += 4 * sizeof(float);
    off = alignup(off, 64);
    __bf16* Xbf     = (__bf16*)(base + off); off += (size_t)MMAX * KDIM * sizeof(__bf16);
    off = alignup(off, 64);
    float* logits   = (float*)(base + off); off += (size_t)MMAX * LDL * sizeof(float);
    (void)ws_size; (void)n_in; (void)out_size;

    // ---- pipeline
    k0_zero_i32<<<(2 * TMAX + 255) / 256, 256, 0, stream>>>(cnt_tot, 2 * TMAX);
    k1_seg_pred<<<nb1, 256, 0, stream>>>(outputs, labels, loss_w, C, HW, BN,
                                         pred_buf, cnt_tot, cnt_hard, partial);
    k2_plan<<<1, 256, 0, stream>>>(cnt_tot, cnt_hard, C, B, nb1, partial,
                                   meta, slot_b, slot_c, slot_kh, slot_ke, segvals);
    k3_select<<<TMAX, 32, 0, stream>>>(labels, pred_buf, HW, meta,
                                       slot_b, slot_c, slot_kh, slot_ke, sample_pix);
    k4_gather<<<MMAX, KDIM, 0, stream>>>(feats, Dch, HW, meta, slot_b, sample_pix, Xbf);
    k5_gemm<<<dim3(MMAX / 16, MMAX / 16), 32, 0, stream>>>(Xbf, logits);
    k6_row<<<MMAX, 256, 0, stream>>>(logits, meta, slot_c, rowvals);
    k7_final<<<1, 256, 0, stream>>>(rowvals, meta, segvals, (float*)d_out);
}
